// ECausalGCN_41721312313869
// MI455X (gfx1250) — matrix operand order repaired and checked by reading, not compiled, not run on hardware
//
#include <hip/hip_runtime.h>
#include <hip/hip_bf16.h>

// ECausalGCN on gfx1250 (MI455X). Heavy GEMMs ([160000x128]x[128x384] etc.)
// run through v_wmma_f32_16x16x32_f16 (wave32 WMMA) with fragment-packed B
// and vectorized A loads; scatter/gather attention phases are bandwidth-bound
// VALU/atomic kernels.

typedef __attribute__((ext_vector_type(16))) _Float16 v16h;
typedef __attribute__((ext_vector_type(8)))  _Float16 v8h;
typedef __attribute__((ext_vector_type(8)))  float    v8f;

static constexpr int NN = 10000;     // nodes
static constexpr int NE = 160000;    // edges
static constexpr int NH = 384;       // HEADS*HID
static constexpr int NG = 64;        // graphs

// ---------------------------------------------------------------- utilities
__global__ void k_fill(float* __restrict__ p, float v, int n) {
  int i = blockIdx.x * blockDim.x + threadIdx.x;
  if (i < n) p[i] = v;
}

__global__ void k_relu(float* __restrict__ p, int n) {
  int i = blockIdx.x * blockDim.x + threadIdx.x;
  if (i < n) { float v = p[i]; p[i] = v > 0.f ? v : 0.f; }
}

__global__ void k_add(const float* __restrict__ a, const float* __restrict__ b,
                      float* __restrict__ c, int n) {
  int i = blockIdx.x * blockDim.x + threadIdx.x;
  if (i < n) c[i] = a[i] + b[i];
}

__global__ void k_cvt16(const float* __restrict__ x, _Float16* __restrict__ y, int n) {
  int i = blockIdx.x * blockDim.x + threadIdx.x;
  if (i < n) y[i] = (_Float16)x[i];
}

// ------------------------------------------------------------- batch norm
// One block per feature column (128 columns), reduce over rows.
__global__ void k_bn_stats(const float* __restrict__ x, int rows,
                           float* __restrict__ mu, float* __restrict__ istd) {
  __shared__ float s1[256], s2[256];
  int c = blockIdx.x;            // 0..127
  int t = threadIdx.x;
  float a = 0.f, b = 0.f;
  for (int r = t; r < rows; r += blockDim.x) {
    float v = x[(size_t)r * 128 + c];
    a += v; b += v * v;
  }
  s1[t] = a; s2[t] = b; __syncthreads();
  for (int s = 128; s > 0; s >>= 1) {
    if (t < s) { s1[t] += s1[t + s]; s2[t] += s2[t + s]; }
    __syncthreads();
  }
  if (t == 0) {
    float m = s1[0] / (float)rows;
    float var = s2[0] / (float)rows - m * m;
    mu[c] = m;
    istd[c] = rsqrtf(var + 1e-5f);
  }
}

__global__ void k_bn_apply16(const float* __restrict__ x, const float* __restrict__ mu,
                             const float* __restrict__ istd, _Float16* __restrict__ y, int n) {
  int i = blockIdx.x * blockDim.x + threadIdx.x;
  if (i < n) {
    int c = i & 127;
    y[i] = (_Float16)((x[i] - mu[c]) * istd[c] + 1e-4f);
  }
}

// --------------------------------------------------------- B fragment pack
// Rewrites B[K,N] (f32, row-major) into per-lane WMMA B-fragment layout:
// Bp[(tile*KSTEPS + step)*32 + lane][16 halves], zero-padded past K.
// Lane layout per CDNA5 ISA: col = lane&15, K = step*32 + (lane>>4)*8 + (j<8 ? j : 8+j).
__global__ void k_pack_b(const float* __restrict__ B, _Float16* __restrict__ Bp,
                         int K, int N) {
  int ksteps = (K + 31) / 32;
  int total = (N / 16) * ksteps * 32;
  int i = blockIdx.x * blockDim.x + threadIdx.x;
  if (i >= total) return;
  int lane = i & 31;
  int step = (i >> 5) % ksteps;
  int tile = i / (32 * ksteps);
  int col  = lane & 15;
  int boff = (lane >> 4) * 8;
  _Float16* dst = Bp + (size_t)i * 16;
#pragma unroll
  for (int j = 0; j < 16; ++j) {
    int kk = step * 32 + boff + (j < 8 ? j : 8 + j);
    dst[j] = (kk < K) ? (_Float16)B[(size_t)kk * N + tile * 16 + col] : (_Float16)0.0f;
  }
}

// --------------------------------------------------------------- WMMA GEMM
// C[M,N] = A[M,K](f16,row-major) x B(packed frags) + bias[N].
// One wave computes a 16x32 C strip (two 16x16 tiles sharing the A fragment);
// 8 waves / 256-thread block cover 128 rows x 32 cols.  K compile-time.
// Requires: M % 16 == 0, N % 32 == 0 (all call sites satisfy this).
template <int K>
__global__ void k_gemm_wmma(const _Float16* __restrict__ A, const v16h* __restrict__ Bp,
                            const float* __restrict__ bias, float* __restrict__ C,
                            int M, int N) {
  constexpr int KSTEPS = (K + 31) / 32;
  const int lane = threadIdx.x & 31;
  const int wave = threadIdx.x >> 5;                 // 0..7
  const int t0 = blockIdx.x * 2;                     // first of two n-tiles
  const int n0 = t0 * 16;
  const int m0 = (blockIdx.y * 8 + wave) * 16;
  if (m0 >= M) return;                               // wave-uniform (EXEC all-ones at WMMA)
  const int aoff = (lane >> 4) * 8;                  // A fragment K split
  const int row  = lane & 15;
  const int col  = lane & 15;

  const _Float16* Arow = A + (size_t)(m0 + row) * K + aoff;
  const v16h* B0 = Bp + (size_t)t0 * KSTEPS * 32 + lane;
  const v16h* B1 = B0 + (size_t)KSTEPS * 32;

  v8f acc0 = {}, acc1 = {};
#pragma unroll
  for (int s = 0; s < KSTEPS; ++s) {
    const int k0 = s * 32;
    if constexpr (K >= 128) __builtin_prefetch(Arow + k0 + 64, 0, 3);  // global_prefetch_b8
    v8h lo = *(const v8h*)(Arow + k0);               // halves k0+aoff .. +7
    v8h hi = {};
    if constexpr (K % 32 == 0) hi = *(const v8h*)(Arow + k0 + 16);     // k0+16+aoff ..
    v16h a = __builtin_shufflevector(lo, hi, 0, 1, 2, 3, 4, 5, 6, 7,
                                     8, 9, 10, 11, 12, 13, 14, 15);
    v16h b0 = B0[s * 32];
    v16h b1 = B1[s * 32];
    acc0 = __builtin_amdgcn_wmma_f32_16x16x32_f16(false, a, false, b0, (short)0,
                                                  acc0, false, false);
    acc1 = __builtin_amdgcn_wmma_f32_16x16x32_f16(false, a, false, b1, (short)0,
                                                  acc1, false, false);
  }
  const int mbase = (lane >> 4) * 8;                 // C: lanes 16-31 hold M=8..15
  float bv0 = bias ? bias[n0 + col] : 0.f;
  float bv1 = bias ? bias[n0 + 16 + col] : 0.f;
#pragma unroll
  for (int i = 0; i < 8; ++i) {
    size_t r = (size_t)(m0 + mbase + i) * N;
    C[r + n0 + col]      = acc0[i] + bv0;
    C[r + n0 + 16 + col] = acc1[i] + bv1;
  }
}

// ------------------------------------------------------------- attention
// logits[e,h] = leaky_relu( hs.a0 + hd.a1 + ep.a2 ), wave per edge.
__global__ void k_logits(const float* __restrict__ h, const float* __restrict__ ep,
                         const int* __restrict__ row, const int* __restrict__ col,
                         const float* __restrict__ a,        // [3,3,128]
                         float* __restrict__ logits,         // [E,3]
                         float* __restrict__ eattn, int store_eattn, int E) {
  int e = blockIdx.x * 8 + (threadIdx.x >> 5);
  if (e >= E) return;
  int lane = threadIdx.x & 31;
  int r = row[e], c = col[e];
  float lsum = 0.f;
  for (int hh = 0; hh < 3; ++hh) {
    float p = 0.f;
    for (int d = lane; d < 128; d += 32) {
      float hs = h[(size_t)r * NH + hh * 128 + d];
      float hd = h[(size_t)c * NH + hh * 128 + d];
      float ev = ep[(size_t)e * NH + hh * 128 + d];
      p += hs * a[hh * 128 + d] + hd * a[384 + hh * 128 + d] + ev * a[768 + hh * 128 + d];
    }
    for (int off = 16; off; off >>= 1) p += __shfl_xor(p, off, 32);
    float l = p > 0.f ? p : 0.2f * p;
    if (lane == 0) logits[(size_t)e * 3 + hh] = l;
    lsum += l;
  }
  if (store_eattn && lane == 0)
    eattn[e] = 1.f / (1.f + __expf(-lsum * (1.f / 3.f)));
}

__device__ inline void atomicMaxF(float* addr, float val) {
  unsigned int* ua = (unsigned int*)addr;
  unsigned int old = *ua;
  while (__uint_as_float(old) < val) {
    unsigned int assumed = old;
    old = atomicCAS(ua, assumed, __float_as_uint(val));
    if (old == assumed) break;
  }
}

__global__ void k_segmax(const float* __restrict__ logits, const int* __restrict__ col,
                         float* __restrict__ m, int E) {
  int i = blockIdx.x * blockDim.x + threadIdx.x;
  if (i >= E * 3) return;
  int e = i / 3, hh = i - e * 3;
  atomicMaxF(&m[(size_t)col[e] * 3 + hh], logits[i]);
}

__global__ void k_expw(const float* __restrict__ logits, const int* __restrict__ col,
                       const float* __restrict__ m, float* __restrict__ w,
                       float* __restrict__ denom, int E) {
  int i = blockIdx.x * blockDim.x + threadIdx.x;
  if (i >= E * 3) return;
  int e = i / 3, hh = i - e * 3;
  float v = __expf(logits[i] - m[(size_t)col[e] * 3 + hh]);
  w[i] = v;
  atomicAdd(&denom[(size_t)col[e] * 3 + hh], v);
}

// x_out[col] += alpha*(hs+ep)/H ; e_out = mean_h(hs+hd+ep). Wave per edge.
__global__ void k_aggr(const float* __restrict__ h, const float* __restrict__ ep,
                       const float* __restrict__ w, const float* __restrict__ denom,
                       const int* __restrict__ row, const int* __restrict__ col,
                       float* __restrict__ xout, float* __restrict__ eout,
                       int store_eout, int E) {
  int e = blockIdx.x * 8 + (threadIdx.x >> 5);
  if (e >= E) return;
  int lane = threadIdx.x & 31;
  int r = row[e], c = col[e];
  float eo[4] = {0.f, 0.f, 0.f, 0.f};
  for (int hh = 0; hh < 3; ++hh) {
    float alpha = w[(size_t)e * 3 + hh] / (denom[(size_t)c * 3 + hh] + 1e-16f);
#pragma unroll
    for (int q = 0; q < 4; ++q) {
      int d = lane + q * 32;
      float hs = h[(size_t)r * NH + hh * 128 + d];
      float hd = h[(size_t)c * NH + hh * 128 + d];
      float ev = ep[(size_t)e * NH + hh * 128 + d];
      atomicAdd(&xout[(size_t)c * 128 + d], alpha * (hs + ev) * (1.f / 3.f));
      eo[q] += hs + hd + ev;
    }
  }
  if (store_eout) {
#pragma unroll
    for (int q = 0; q < 4; ++q)
      eout[(size_t)e * 128 + lane + q * 32] = eo[q] * (1.f / 3.f);
  }
}

// ------------------------------------------------------- gating / pooling
__global__ void k_natt(const float* __restrict__ x, const float* __restrict__ W,
                       const float* __restrict__ b, float* __restrict__ xc,
                       float* __restrict__ xo, int N) {
  int n = blockIdx.x * 8 + (threadIdx.x >> 5);
  if (n >= N) return;
  int lane = threadIdx.x & 31;
  float p0 = 0.f, p1 = 0.f;
  for (int d = lane; d < 128; d += 32) {
    float v = x[(size_t)n * 128 + d];
    p0 += v * W[d * 2];
    p1 += v * W[d * 2 + 1];
  }
  for (int off = 16; off; off >>= 1) {
    p0 += __shfl_xor(p0, off, 32);
    p1 += __shfl_xor(p1, off, 32);
  }
  float l0 = p0 + b[0], l1 = p1 + b[1];
  float mx = fmaxf(l0, l1);
  float e0 = __expf(l0 - mx), e1 = __expf(l1 - mx);
  float inv = 1.f / (e0 + e1);
  float a0 = e0 * inv, a1 = e1 * inv;
  for (int d = lane; d < 128; d += 32) {
    float v = x[(size_t)n * 128 + d];
    xc[(size_t)n * 128 + d] = a0 * v;
    xo[(size_t)n * 128 + d] = a1 * v;
  }
}

__global__ void k_edge_split(const float* __restrict__ e, const float* __restrict__ eattn,
                             float* __restrict__ dst, int one_minus, int n) {
  int i = blockIdx.x * blockDim.x + threadIdx.x;
  if (i >= n) return;
  float t = eattn[i >> 7];
  if (one_minus) t = 1.f - t;
  dst[i] = t * e[i];
}

__global__ void k_graph_sum(const float* __restrict__ x, const int* __restrict__ batch,
                            float* __restrict__ g, int N) {
  int i = blockIdx.x * blockDim.x + threadIdx.x;
  if (i >= N * 128) return;
  int n = i >> 7, d = i & 127;
  float v = x[i];
  v = v > 0.f ? v : 0.f;                      // relu before segment_sum
  atomicAdd(&g[(size_t)batch[n] * 128 + d], v);
}

// fused: bn(x) @ W2 + b2 -> log_softmax.  Wave per graph, lanes 0..9 = classes.
__global__ void k_fc2_lsm(const float* __restrict__ x, const float* __restrict__ mu,
                          const float* __restrict__ istd, const float* __restrict__ W2,
                          const float* __restrict__ b2, float* __restrict__ out, int G) {
  int g = blockIdx.x * 8 + (threadIdx.x >> 5);
  if (g >= G) return;
  int lane = threadIdx.x & 31;
  float z = -1e30f;
  if (lane < 10) {
    float s = b2[lane];
    for (int d = 0; d < 128; ++d) {
      float xv = (x[(size_t)g * 128 + d] - mu[d]) * istd[d] + 1e-4f;
      s += xv * W2[d * 10 + lane];
    }
    z = s;
  }
  float mx = z;
  for (int off = 16; off; off >>= 1) mx = fmaxf(mx, __shfl_xor(mx, off, 32));
  float ex = (lane < 10) ? __expf(z - mx) : 0.f;
  float sum = ex;
  for (int off = 16; off; off >>= 1) sum += __shfl_xor(sum, off, 32);
  if (lane < 10) out[(size_t)g * 10 + lane] = (z - mx) - __logf(sum);
}

// ================================================================= launch
extern "C" void kernel_launch(void* const* d_in, const int* in_sizes, int n_in,
                              void* d_out, int out_size, void* d_ws, size_t ws_size,
                              hipStream_t stream) {
  (void)in_sizes; (void)n_in; (void)out_size; (void)ws_size;
  const float* x     = (const float*)d_in[0];
  const float* eattr = (const float*)d_in[1];
  const float* Wn0   = (const float*)d_in[2];
  const float* We0   = (const float*)d_in[3];
  const float* b0    = (const float*)d_in[4];
  const float* a0    = (const float*)d_in[5];
  const float* Wn    = (const float*)d_in[6];
  const float* We    = (const float*)d_in[7];
  const float* bb    = (const float*)d_in[8];
  const float* aa    = (const float*)d_in[9];
  const float* nattW = (const float*)d_in[10];
  const float* nattb = (const float*)d_in[11];
  const float* fcW1  = (const float*)d_in[12];
  const float* fcb1  = (const float*)d_in[13];
  const float* fcW2  = (const float*)d_in[14];
  const float* fcb2  = (const float*)d_in[15];
  const int*   eidx  = (const int*)d_in[16];
  const int*   batch = (const int*)d_in[17];
  float* out = (float*)d_out;
  const int* row = eidx;
  const int* col = eidx + NE;

  // ---- workspace bump allocator
  size_t off = 0;
  auto alloc = [&](size_t bytes) {
    size_t a = (off + 255) & ~(size_t)255;
    off = a + bytes;
    return (void*)((char*)d_ws + a);
  };
  // packed-B fragment sizes (halves): (N/16) * KSTEPS * 32 * 16
  const size_t SZ_B128_384 = (size_t)24 * 4 * 32 * 16;   // K=128, N=384
  const size_t SZ_B16_384  = (size_t)24 * 1 * 32 * 16;   // K=16 (padded), N=384
  const size_t SZ_B128_128 = (size_t)8 * 4 * 32 * 16;    // K=128, N=128
  _Float16* Wn0p = (_Float16*)alloc(SZ_B128_384 * 2);
  _Float16* We0p = (_Float16*)alloc(SZ_B16_384 * 2);
  _Float16* Wnp  = (_Float16*)alloc(5 * SZ_B128_384 * 2);
  _Float16* Wep  = (_Float16*)alloc(5 * SZ_B128_384 * 2);
  _Float16* W1p  = (_Float16*)alloc(3 * SZ_B128_128 * 2);
  _Float16* ea16 = (_Float16*)alloc((size_t)NE * 16 * 2);
  _Float16* xh16 = (_Float16*)alloc((size_t)NN * 128 * 2);
  _Float16* eh16 = (_Float16*)alloc((size_t)NE * 128 * 2);
  float* hbuf  = (float*)alloc((size_t)NN * NH * 4);
  float* epbuf = (float*)alloc((size_t)NE * NH * 4);
  float* logit = (float*)alloc((size_t)NE * 3 * 4);
  float* wbuf  = (float*)alloc((size_t)NE * 3 * 4);
  float* mbuf  = (float*)alloc((size_t)NN * 3 * 4);
  float* dbuf  = (float*)alloc((size_t)NN * 3 * 4);
  float* xout  = (float*)alloc((size_t)NN * 128 * 4);
  float* eout  = (float*)alloc((size_t)NE * 128 * 4);
  float* eatb  = (float*)alloc((size_t)NE * 4);
  float* xc    = (float*)alloc((size_t)NN * 128 * 4);
  float* xo    = (float*)alloc((size_t)NN * 128 * 4);
  float* etmp  = (float*)alloc((size_t)NE * 128 * 4);
  float* xcg   = (float*)alloc((size_t)NG * 128 * 4);
  float* xog   = (float*)alloc((size_t)NG * 128 * 4);
  float* gtmp  = (float*)alloc((size_t)NG * 128 * 4);
  float* gh    = (float*)alloc((size_t)NG * 128 * 4);
  _Float16* g16 = (_Float16*)alloc((size_t)NG * 128 * 2);
  float* mu1 = (float*)alloc(128 * 4);
  float* is1 = (float*)alloc(128 * 4);
  float* mu2 = (float*)alloc(128 * 4);
  float* is2 = (float*)alloc(128 * 4);

  auto cdiv = [](int a, int b) { return (a + b - 1) / b; };
  auto cvt = [&](const float* s, _Float16* d, int n) {
    k_cvt16<<<cdiv(n, 256), 256, 0, stream>>>(s, d, n);
  };
  auto fill = [&](float* p, float v, int n) {
    k_fill<<<cdiv(n, 256), 256, 0, stream>>>(p, v, n);
  };
  auto packB = [&](const float* B, _Float16* Bp, int K, int N) {
    int total = (N / 16) * ((K + 31) / 32) * 32;
    k_pack_b<<<cdiv(total, 256), 256, 0, stream>>>(B, Bp, K, N);
  };
  auto gemm128 = [&](const _Float16* A, const _Float16* Bp, const float* bias, float* C,
                     int M, int N) {
    k_gemm_wmma<128><<<dim3(N / 32, cdiv(M, 128)), 256, 0, stream>>>(
        A, (const v16h*)Bp, bias, C, M, N);
  };
  auto gemm16 = [&](const _Float16* A, const _Float16* Bp, const float* bias, float* C,
                    int M, int N) {
    k_gemm_wmma<16><<<dim3(N / 32, cdiv(M, 128)), 256, 0, stream>>>(
        A, (const v16h*)Bp, bias, C, M, N);
  };
  auto bn16 = [&](const float* src, int rows, _Float16* dst) {
    k_bn_stats<<<128, 256, 0, stream>>>(src, rows, mu1, is1);
    k_bn_apply16<<<cdiv(rows * 128, 256), 256, 0, stream>>>(src, mu1, is1, dst, rows * 128);
  };
  const int EW = cdiv(NE, 8);   // wave-per-edge grids
  const int NW = cdiv(NN, 8);
  auto egat_attn = [&](const float* ap, float* xdst, float* edst, int store_eout,
                       int store_eattn) {
    k_logits<<<EW, 256, 0, stream>>>(hbuf, epbuf, row, col, ap, logit, eatb, store_eattn, NE);
    fill(mbuf, -1e30f, NN * 3);
    k_segmax<<<cdiv(NE * 3, 256), 256, 0, stream>>>(logit, col, mbuf, NE);
    fill(dbuf, 0.f, NN * 3);
    k_expw<<<cdiv(NE * 3, 256), 256, 0, stream>>>(logit, col, mbuf, wbuf, dbuf, NE);
    fill(xdst, 0.f, NN * 128);
    k_aggr<<<EW, 256, 0, stream>>>(hbuf, epbuf, wbuf, dbuf, row, col, xdst, edst,
                                   store_eout, NE);
  };

  // ---- pack weights into WMMA B fragments (f32 -> f16) once per call
  packB(Wn0, Wn0p, 128, NH);
  packB(We0, We0p, 16, NH);
  for (int i = 0; i < 5; ++i) {
    packB(Wn + (size_t)i * 128 * NH, Wnp + (size_t)i * SZ_B128_384, 128, NH);
    packB(We + (size_t)i * 128 * NH, Wep + (size_t)i * SZ_B128_384, 128, NH);
  }
  for (int j = 0; j < 3; ++j)
    packB(fcW1 + (size_t)j * 128 * 128, W1p + (size_t)j * SZ_B128_128, 128, 128);
  cvt(eattr, ea16, NE * 16);

  // ---- layer 0
  bn16(x, NN, xh16);
  gemm128(xh16, Wn0p, b0, hbuf, NN, NH);
  gemm16(ea16, We0p, b0 + 384, epbuf, NE, NH);
  egat_attn(a0, xout, eout, 1, 0);
  k_relu<<<cdiv(NN * 128, 256), 256, 0, stream>>>(xout, NN * 128);
  k_relu<<<cdiv(NE * 128, 256), 256, 0, stream>>>(eout, NE * 128);

  // ---- 3 conv layers
  for (int i = 0; i < 3; ++i) {
    bn16(xout, NN, xh16);
    cvt(eout, eh16, NE * 128);
    gemm128(xh16, Wnp + (size_t)i * SZ_B128_384, bb + i * 768, hbuf, NN, NH);
    gemm128(eh16, Wep + (size_t)i * SZ_B128_384, bb + i * 768 + 384, epbuf, NE, NH);
    egat_attn(aa + i * 1152, xout, eout, 1, i == 2);
    k_relu<<<cdiv(NN * 128, 256), 256, 0, stream>>>(xout, NN * 128);
    k_relu<<<cdiv(NE * 128, 256), 256, 0, stream>>>(eout, NE * 128);
  }

  // ---- node attention gate
  k_natt<<<NW, 256, 0, stream>>>(xout, nattW, nattb, xc, xo, NN);

  // ---- causal branch (layer 3)
  k_edge_split<<<cdiv(NE * 128, 256), 256, 0, stream>>>(eout, eatb, etmp, 0, NE * 128);
  bn16(xc, NN, xh16);
  bn16(etmp, NE, eh16);
  gemm128(xh16, Wnp + (size_t)3 * SZ_B128_384, bb + 3 * 768, hbuf, NN, NH);
  gemm128(eh16, Wep + (size_t)3 * SZ_B128_384, bb + 3 * 768 + 384, epbuf, NE, NH);
  egat_attn(aa + 3 * 1152, xc, etmp, 0, 0);
  fill(xcg, 0.f, NG * 128);
  k_graph_sum<<<cdiv(NN * 128, 256), 256, 0, stream>>>(xc, batch, xcg, NN);

  // ---- other branch (layer 4)
  k_edge_split<<<cdiv(NE * 128, 256), 256, 0, stream>>>(eout, eatb, etmp, 1, NE * 128);
  bn16(xo, NN, xh16);
  bn16(etmp, NE, eh16);
  gemm128(xh16, Wnp + (size_t)4 * SZ_B128_384, bb + 4 * 768, hbuf, NN, NH);
  gemm128(eh16, Wep + (size_t)4 * SZ_B128_384, bb + 4 * 768 + 384, epbuf, NE, NH);
  egat_attn(aa + 4 * 1152, xo, etmp, 0, 0);
  fill(xog, 0.f, NG * 128);
  k_graph_sum<<<cdiv(NN * 128, 256), 256, 0, stream>>>(xo, batch, xog, NN);

  // ---- readouts
  auto readout = [&](const float* gin, const _Float16* W1, const float* b1,
                     const float* W2, const float* b2, float* op) {
    k_bn_stats<<<128, 256, 0, stream>>>(gin, NG, mu1, is1);
    k_bn_apply16<<<cdiv(NG * 128, 256), 256, 0, stream>>>(gin, mu1, is1, g16, NG * 128);
    gemm128(g16, W1, b1, gh, NG, 128);
    k_relu<<<cdiv(NG * 128, 256), 256, 0, stream>>>(gh, NG * 128);
    k_bn_stats<<<128, 256, 0, stream>>>(gh, NG, mu2, is2);
    k_fc2_lsm<<<cdiv(NG, 8), 256, 0, stream>>>(gh, mu2, is2, W2, b2, op, NG);
  };
  readout(xcg, W1p,                  fcb1,       fcW2,        fcb2,      out);
  readout(xog, W1p + SZ_B128_128,    fcb1 + 128, fcW2 + 1280, fcb2 + 10, out + NG * 10);
  k_add<<<cdiv(NG * 128, 256), 256, 0, stream>>>(xcg, xog, gtmp, NG * 128);
  readout(gtmp, W1p + 2 * SZ_B128_128, fcb1 + 256, fcW2 + 2560, fcb2 + 20, out + 2 * NG * 10);
}
